// CompMLP_28664611733761
// MI455X (gfx1250) — matrix operand order repaired and checked by reading, not compiled, not run on hardware
//
#include <hip/hip_runtime.h>

typedef __attribute__((ext_vector_type(16))) _Float16 v16h;
typedef __attribute__((ext_vector_type(8)))  float    v8f;

#define B_NC   171
#define ZS     288   // 272 padded to 9 k-tiles of 32
#define H1DIM  256
#define H2DIM  128
#define ROWS   64    // batch rows per workgroup

// ---- weight prep: fp32 -> fp16, transposed (col-major of W), K padded ----
__global__ __launch_bounds__(256) void prep_weights(
    const float* __restrict__ W1, const float* __restrict__ W2,
    _Float16* __restrict__ w1t, _Float16* __restrict__ w2t) {
  int i = blockIdx.x * 256 + threadIdx.x;
  if (i < 256 * ZS) {                      // w1t[n][k], n in [0,256), k in [0,288)
    int n = i / ZS, k = i % ZS;
    w1t[i] = (k < 272) ? (_Float16)W1[k * 256 + n] : (_Float16)0.f;
  } else {
    int j = i - 256 * ZS;                  // w2t[n][k], n in [0,128), k in [0,256)
    if (j < 128 * 256) {
      int n = j / 256, k = j % 256;
      w2t[j] = (_Float16)W2[k * 128 + n];
    }
  }
}

__device__ __forceinline__ int fixi(int i, int n) { return i < 0 ? n - 1 : i; }

// A fragment (16x32 f16): lane<16 holds row=laneLo, K {0..15}; lane>=16 K {16..31}
__device__ __forceinline__ v16h load_a(const _Float16* rowp, int kk, int laneHi) {
  union { v16h v; uint4 q[2]; } f;
  f.q[0] = *(const uint4*)(rowp + kk + laneHi * 16);
  f.q[1] = *(const uint4*)(rowp + kk + laneHi * 16 + 8);
  return f.v;
}

__global__ __launch_bounds__(256) void comp_mlp(
    const int* __restrict__ my_idx, const int* __restrict__ ally,
    const int* __restrict__ enem, const int* __restrict__ misc_idx,
    const float* __restrict__ embc,
    const float* __restrict__ emb_sp, const float* __restrict__ emb_pri,
    const float* __restrict__ emb_sub, const float* __restrict__ emb_key,
    const float* __restrict__ emb_pat,
    const _Float16* __restrict__ w1t, const float* __restrict__ b1,
    const _Float16* __restrict__ w2t, const float* __restrict__ b2,
    const float* __restrict__ W3, const float* __restrict__ b3,
    float* __restrict__ out, int nB) {
  __shared__ _Float16 sm[26624] __attribute__((aligned(16)));   // 52 KB
  _Float16* zs  = sm;              // 64x288 f16 (36 KB); reused as h1 (64x256)
  _Float16* h1s = sm;
  _Float16* wsl = sm + ROWS * ZS;  // 8192 halves (16 KB): weight slab; reused as h2 (64x128)
  _Float16* h2s = wsl;

  const int tid = threadIdx.x;
  const int lane = tid & 31, wave = tid >> 5;
  const int laneLo = lane & 15, laneHi = lane >> 4;
  const int row0 = blockIdx.x * ROWS;

  // ---------------- gather: z[64][288] in f16 ----------------
  {
    const int rl = tid >> 2, sub = tid & 3;   // 4 threads per row
    const int g = row0 + rl;
    if (g < nB) {
      int myi = fixi(my_idx[g], B_NC);
      int a0 = fixi(ally[g*4+0], B_NC), a1 = fixi(ally[g*4+1], B_NC),
          a2 = fixi(ally[g*4+2], B_NC), a3 = fixi(ally[g*4+3], B_NC);
      int e0 = fixi(enem[g*5+0], B_NC), e1 = fixi(enem[g*5+1], B_NC),
          e2 = fixi(enem[g*5+2], B_NC), e3 = fixi(enem[g*5+3], B_NC),
          e4 = fixi(enem[g*5+4], B_NC);
      int m0 = misc_idx[g*5+0], m1 = misc_idx[g*5+1], m2 = misc_idx[g*5+2],
          m3 = misc_idx[g*5+3], m4 = misc_idx[g*5+4];
      for (int c = sub; c < ZS; c += 4) {
        float v;
        if (c < 64) {
          v = embc[myi*64 + c];
        } else if (c < 128) {
          int cc = c - 64;
          v = embc[a0*64+cc] + embc[a1*64+cc] + embc[a2*64+cc] + embc[a3*64+cc];
        } else if (c < 192) {
          int cc = c - 128;
          v = embc[e0*64+cc] + embc[e1*64+cc] + embc[e2*64+cc] +
              embc[e3*64+cc] + embc[e4*64+cc];
        } else if (c < 272) {
          int cc = c - 192, j = cc >> 4, o = cc & 15;
          const float* t; int mi, nv;
          switch (j) {
            case 0:  t = emb_sp;  mi = m0; nv = 33; break;
            case 1:  t = emb_pri; mi = m1; nv = 9;  break;
            case 2:  t = emb_sub; mi = m2; nv = 9;  break;
            case 3:  t = emb_key; mi = m3; nv = 65; break;
            default: t = emb_pat; mi = m4; nv = 65; break;
          }
          v = t[fixi(mi, nv)*16 + o];
        } else {
          v = 0.f;
        }
        zs[rl*ZS + c] = (_Float16)v;
      }
    } else {
      for (int c = sub; c < ZS; c += 4) zs[rl*ZS + c] = (_Float16)0.f;
    }
  }

  // ---------------- layer 1: h1 = relu(z @ W1 + b1) ----------------
  v8f acc[8] = {};                     // [mt 0..3][ct 0..1]
  for (int kt = 0; kt < 9; ++kt) {
    __syncthreads();                   // gather done / prev slab consumed
    {                                  // stage W1t k-slab: wsl[n*32+k], n = tid
      const uint4* s = (const uint4*)(w1t + tid * ZS + kt * 32);
      uint4* d = (uint4*)(wsl + tid * 32);
      d[0] = s[0]; d[1] = s[1];
    }
    __syncthreads();
    v16h a0 = load_a(zs + (0*16 + laneLo)*ZS, kt*32, laneHi);
    v16h a1 = load_a(zs + (1*16 + laneLo)*ZS, kt*32, laneHi);
    v16h a2 = load_a(zs + (2*16 + laneLo)*ZS, kt*32, laneHi);
    v16h a3 = load_a(zs + (3*16 + laneLo)*ZS, kt*32, laneHi);
    #pragma unroll
    for (int ct = 0; ct < 2; ++ct) {
      int nt = wave * 2 + ct;
      v16h b = *(const v16h*)(wsl + (nt*16 + laneLo)*32 + laneHi*16);
      acc[0*2+ct] = __builtin_amdgcn_wmma_f32_16x16x32_f16(false, a0, false, b, (short)0, acc[0*2+ct], false, false);
      acc[1*2+ct] = __builtin_amdgcn_wmma_f32_16x16x32_f16(false, a1, false, b, (short)0, acc[1*2+ct], false, false);
      acc[2*2+ct] = __builtin_amdgcn_wmma_f32_16x16x32_f16(false, a2, false, b, (short)0, acc[2*2+ct], false, false);
      acc[3*2+ct] = __builtin_amdgcn_wmma_f32_16x16x32_f16(false, a3, false, b, (short)0, acc[3*2+ct], false, false);
    }
  }
  __syncthreads();                     // all z reads done -> safe to alias as h1
  #pragma unroll
  for (int ct = 0; ct < 2; ++ct) {
    int col = (wave*2 + ct)*16 + laneLo;
    float bias = b1[col];
    #pragma unroll
    for (int mt = 0; mt < 4; ++mt) {
      #pragma unroll
      for (int r = 0; r < 8; ++r) {
        float v = acc[mt*2+ct][r] + bias;
        h1s[(mt*16 + laneHi*8 + r)*H1DIM + col] = (_Float16)fmaxf(v, 0.f);
      }
    }
  }

  // ---------------- layer 2: h2 = relu(h1 @ W2 + b2) ----------------
  v8f acc2[4] = {};
  for (int kt = 0; kt < 8; ++kt) {
    __syncthreads();                   // h1 visible / prev slab consumed
    { int n = tid >> 1, h = tid & 1;   // stage W2t k-slab (128 rows x 32)
      const uint4* s = (const uint4*)(w2t + n * 256 + kt * 32 + h * 16);
      uint4* d = (uint4*)(wsl + n * 32 + h * 16);
      d[0] = s[0]; d[1] = s[1];
    }
    __syncthreads();
    v16h b = *(const v16h*)(wsl + (wave*16 + laneLo)*32 + laneHi*16);
    #pragma unroll
    for (int mt = 0; mt < 4; ++mt) {
      v16h a = load_a(h1s + (mt*16 + laneLo)*H1DIM, kt*32, laneHi);
      acc2[mt] = __builtin_amdgcn_wmma_f32_16x16x32_f16(false, a, false, b, (short)0, acc2[mt], false, false);
    }
  }
  __syncthreads();                     // slab reads done -> safe to alias as h2
  { int col = wave*16 + laneLo;
    float bias = b2[col];
    #pragma unroll
    for (int mt = 0; mt < 4; ++mt) {
      #pragma unroll
      for (int r = 0; r < 8; ++r) {
        float v = acc2[mt][r] + bias;
        h2s[(mt*16 + laneHi*8 + r)*H2DIM + col] = (_Float16)fmaxf(v, 0.f);
      }
    }
  }
  __syncthreads();

  // ---------------- layer 3: out = h2 @ W3 + b3 ----------------
  if (tid < ROWS) {
    int g = row0 + tid;
    if (g < nB) {
      const _Float16* hp = h2s + tid * H2DIM;
      float s = 0.f;
      #pragma unroll 8
      for (int j = 0; j < H2DIM; ++j) s += (float)hp[j] * W3[j];
      out[g] = s + b3[0];
    }
  }
}

extern "C" void kernel_launch(void* const* d_in, const int* in_sizes, int n_in,
                              void* d_out, int out_size, void* d_ws, size_t ws_size,
                              hipStream_t stream) {
  const int*   my_idx = (const int*)d_in[0];
  const int*   ally   = (const int*)d_in[1];
  const int*   enem   = (const int*)d_in[2];
  const int*   misc   = (const int*)d_in[3];
  const float* embc   = (const float*)d_in[4];
  const float* esp    = (const float*)d_in[5];
  const float* epri   = (const float*)d_in[6];
  const float* esub   = (const float*)d_in[7];
  const float* ekey   = (const float*)d_in[8];
  const float* epat   = (const float*)d_in[9];
  const float* W1     = (const float*)d_in[10];
  const float* b1     = (const float*)d_in[11];
  const float* W2     = (const float*)d_in[12];
  const float* b2     = (const float*)d_in[13];
  const float* W3     = (const float*)d_in[14];
  const float* b3     = (const float*)d_in[15];
  float* out = (float*)d_out;
  const int nB = in_sizes[0];

  _Float16* w1t = (_Float16*)d_ws;          // 256*288 halves
  _Float16* w2t = w1t + 256 * ZS;           // 128*256 halves (total ~208 KB)

  const int prep_elems = 256 * ZS + 128 * 256;
  prep_weights<<<(prep_elems + 255) / 256, 256, 0, stream>>>(W1, W2, w1t, w2t);

  const int ngrp = (nB + ROWS - 1) / ROWS;
  comp_mlp<<<ngrp, 256, 0, stream>>>(my_idx, ally, enem, misc, embc,
                                     esp, epri, esub, ekey, epat,
                                     w1t, b1, w2t, b2, W3, b3, out, nB);
}
